// EmbGNN_43911745634882
// MI455X (gfx1250) — matrix operand order repaired and checked by reading, not compiled, not run on hardware
//
#include <hip/hip_runtime.h>

// ---------------------------------------------------------------------------
// Types for CDNA5 WMMA (wave32, v_wmma_f32_16x16x32_bf16)
// ---------------------------------------------------------------------------
typedef __bf16 bf16x16 __attribute__((ext_vector_type(16)));
typedef float  f32x8   __attribute__((ext_vector_type(8)));

#define GN      50000   // nodes
#define GE      800000  // edges
#define INDIM   128
#define EMBDIM  64
#define HID     128
#define OUTDIM  64
#define NGRAPH  128

__device__ __forceinline__ unsigned short f2bf(float f) {
    unsigned int u = __float_as_uint(f);
    u += 0x7FFFu + ((u >> 16) & 1u);          // round-to-nearest-even
    return (unsigned short)(u >> 16);
}
__device__ __forceinline__ unsigned int pack2bf(float a, float b) {
    return (unsigned int)f2bf(a) | ((unsigned int)f2bf(b) << 16);
}
__device__ __forceinline__ float gelu_tanh(float x) {
    float x3 = x * x * x;
    float t  = tanhf(0.7978845608028654f * (x + 0.044715f * x3));
    return 0.5f * x * (1.0f + t);
}

// ---------------------------------------------------------------------------
// Weight prep: W[K][C] f32  ->  Wt[C][K] bf16  (row-major over C, pairs of K
// contiguous so the B-fragment dword loads are direct)
// ---------------------------------------------------------------------------
__global__ void k_w_to_bf16t(const float* __restrict__ W,
                             unsigned short* __restrict__ Wt, int K, int C) {
    int i = blockIdx.x * blockDim.x + threadIdx.x;
    if (i >= K * C) return;
    int c = i / K, k = i - c * K;
    Wt[(size_t)c * K + k] = f2bf(W[(size_t)k * C + c]);
}

// ---------------------------------------------------------------------------
// Feature prep: h0[n, 0:128] = x[n]/||x[n]||, h0[n, 128:192] = emb[node_index[n]]
// one 128-thread block per node
// ---------------------------------------------------------------------------
__global__ void k_feat(const float* __restrict__ x, const int* __restrict__ nidx,
                       const float* __restrict__ emb, float* __restrict__ h0) {
    __shared__ float red[128];
    int n = blockIdx.x, t = threadIdx.x;
    float v = x[(size_t)n * INDIM + t];
    red[t] = v * v;
    __syncthreads();
    for (int off = 64; off > 0; off >>= 1) {
        if (t < off) red[t] += red[t + off];
        __syncthreads();
    }
    float nrm = sqrtf(red[0]);
    float inv = 1.0f / (nrm == 0.0f ? 1e-8f : nrm);
    h0[(size_t)n * (INDIM + EMBDIM) + t] = v * inv;
    if (t < EMBDIM)
        h0[(size_t)n * (INDIM + EMBDIM) + INDIM + t] =
            emb[(size_t)nidx[n] * EMBDIM + t];
}

// ---------------------------------------------------------------------------
// CSR build: count (init 1 for self-loop), scan, fill
// ---------------------------------------------------------------------------
__global__ void k_cnt_init(int* __restrict__ cnt, int n) {
    int i = blockIdx.x * blockDim.x + threadIdx.x;
    if (i < n) cnt[i] = 1;  // self-loop
}
__global__ void k_cnt_edges(const int* __restrict__ dst, int* __restrict__ cnt, int e) {
    int i = blockIdx.x * blockDim.x + threadIdx.x;
    if (i < e) atomicAdd(&cnt[dst[i]], 1);
}
// single-block exclusive scan; writes rowptr[0..n] and a mutable copy 'next'
// (next may alias the input 'deg' buffer: each element is read before written)
__global__ void k_scan(const int* __restrict__ deg, int* __restrict__ rowptr,
                       int* __restrict__ next, int n) {
    __shared__ int sh[1024];
    int carry = 0;
    for (int base = 0; base < n; base += 1024) {
        int i = base + (int)threadIdx.x;
        int d = (i < n) ? deg[i] : 0;
        sh[threadIdx.x] = d;
        __syncthreads();
        for (int off = 1; off < 1024; off <<= 1) {
            int t = (threadIdx.x >= (unsigned)off) ? sh[threadIdx.x - off] : 0;
            __syncthreads();
            sh[threadIdx.x] += t;
            __syncthreads();
        }
        int incl = sh[threadIdx.x];
        int excl = incl - d + carry;
        if (i < n) { rowptr[i] = excl; next[i] = excl; }
        __syncthreads();
        carry += sh[1023];
        __syncthreads();
    }
    if (threadIdx.x == 0) rowptr[n] = carry;
}
__global__ void k_fill_edges(const int* __restrict__ src, const int* __restrict__ dst,
                             int* __restrict__ next, int* __restrict__ csr, int e) {
    int i = blockIdx.x * blockDim.x + threadIdx.x;
    if (i < e) {
        int pos = atomicAdd(&next[dst[i]], 1);
        csr[pos] = src[i];
    }
}
__global__ void k_fill_self(int* __restrict__ next, int* __restrict__ csr, int n) {
    int i = blockIdx.x * blockDim.x + threadIdx.x;
    if (i < n) {
        int pos = atomicAdd(&next[i], 1);
        csr[pos] = i;
    }
}

// ---------------------------------------------------------------------------
// WMMA GEMM:  Out[M,C] = act( A[M,KDIM](f32) * Wt[C,KDIM](bf16)^T (+bias) )
// All tile counts are compile-time so accumulators stay in VGPRs:
//   KDIM : K dimension (multiple of 32)      -> k-loop count known
//   NCT  : C/16 column tiles (C = 16*NCT)    -> acc[NCT], bv[NCT] register-resident
//   ACT  : 0 = none/no-bias, 2 = bias + leaky_relu(0.01)
// block = 256 threads = 8 waves; block tile = 128 rows x C cols; each wave owns
// 16 rows x C cols. Per k-step: ALL B fragments are preloaded into distinct
// registers first, so the global loads overlap the whole WMMA chain instead of
// serializing load-latency before every WMMA.
// ---------------------------------------------------------------------------
template <int KDIM, int NCT, int ACT>
__global__ __launch_bounds__(256)
void k_gemm_wmma(const float* __restrict__ A, const unsigned int* __restrict__ Wt,
                 const float* __restrict__ bias, float* __restrict__ Out,
                 int M) {
    constexpr int CDIM = NCT * 16;
    constexpr int Kd   = KDIM >> 1;        // dwords per Wt row
    __shared__ unsigned int ldsA[128][17]; // 128 rows x 32 bf16 (16 dwords) + pad
    int tid  = threadIdx.x;
    int wave = tid >> 5, lane = tid & 31;
    int half = lane >> 4, mrow = lane & 15;
    int blockRow = blockIdx.x * 128;

    f32x8 acc[NCT];
    f32x8 zero = {0.f, 0.f, 0.f, 0.f, 0.f, 0.f, 0.f, 0.f};
#pragma unroll
    for (int t = 0; t < NCT; ++t) acc[t] = zero;

    union AV { unsigned int u[8]; bf16x16 v; };

#pragma unroll 1
    for (int k0 = 0; k0 < KDIM; k0 += 32) {
        // cooperative load + f32->bf16 convert of A tile (128 x 32)
#pragma unroll
        for (int p = 0; p < 8; ++p) {
            int row = p * 16 + (tid >> 4);
            int cp  = tid & 15;
            int gr  = blockRow + row;
            float fx = 0.f, fy = 0.f;
            if (gr < M) {
                const float* ap = A + (size_t)gr * KDIM + k0 + cp * 2;
                fx = ap[0]; fy = ap[1];
            }
            ldsA[row][cp] = pack2bf(fx, fy);
        }
        __syncthreads();

        // Preload ALL B fragments (32x16 bf16 tiles: lanes 0-15 K=0..15,
        // lanes 16-31 K=16..31) into distinct registers.
        AV bv[NCT];
#pragma unroll
        for (int t = 0; t < NCT; ++t) {
            int base = (t * 16 + mrow) * Kd + (k0 >> 1) + half * 8;
#pragma unroll
            for (int v = 0; v < 8; ++v) bv[t].u[v] = Wt[base + v];
        }

        // A fragment (16x32 bf16: lanes 0-15 K=0..15, lanes 16-31 K=16..31)
        AV av;
#pragma unroll
        for (int v = 0; v < 8; ++v)
            av.u[v] = ldsA[wave * 16 + mrow][half * 8 + v];

        // WMMA chain (independent accumulators -> back-to-back issue)
#pragma unroll
        for (int t = 0; t < NCT; ++t) {
            acc[t] = __builtin_amdgcn_wmma_f32_16x16x32_bf16(
                false, av.v, false, bv[t].v, (short)0, acc[t], false, false);
        }
        __syncthreads();
    }

    // epilogue (C/D layout: VGPR r -> M = r + 8*half, N = lane&15)
#pragma unroll
    for (int t = 0; t < NCT; ++t) {
        int col  = t * 16 + mrow;
        float bv = (ACT == 2) ? bias[col] : 0.f;
#pragma unroll
        for (int r = 0; r < 8; ++r) {
            int row = blockRow + wave * 16 + half * 8 + r;
            if (row < M) {
                float v = acc[t][r] + bv;
                if (ACT == 2) v = v > 0.f ? v : 0.01f * v;
                Out[(size_t)row * CDIM + col] = v;
            }
        }
    }
}

// ---------------------------------------------------------------------------
// Attention vectors: asn[i] = h[i]·a_s ; adn[i] = h[i]·a_d   (one wave/node)
// ---------------------------------------------------------------------------
__global__ __launch_bounds__(256)
void k_attvec(const float* __restrict__ h, const float* __restrict__ a_s,
              const float* __restrict__ a_d, float* __restrict__ asn,
              float* __restrict__ adn, int n) {
    int node = blockIdx.x * 8 + (threadIdx.x >> 5);
    int lane = threadIdx.x & 31;
    if (node >= n) return;
    const float4 hv = *(const float4*)(h + (size_t)node * HID + lane * 4);
    const float4 sv = *(const float4*)(a_s + lane * 4);
    const float4 dv = *(const float4*)(a_d + lane * 4);
    float ps = hv.x * sv.x + hv.y * sv.y + hv.z * sv.z + hv.w * sv.w;
    float pd = hv.x * dv.x + hv.y * dv.y + hv.z * dv.z + hv.w * dv.w;
    for (int off = 16; off > 0; off >>= 1) {
        ps += __shfl_xor(ps, off, 32);
        pd += __shfl_xor(pd, off, 32);
    }
    if (lane == 0) { asn[node] = ps; adn[node] = pd; }
}

// ---------------------------------------------------------------------------
// GAT aggregation with online softmax over CSR in-edges; one wave per dst.
// out[n] = gelu( sum_e softmax(lrelu(asn[src]+adn[n],0.2)) * h[src] + bias )
// ---------------------------------------------------------------------------
__global__ __launch_bounds__(256)
void k_agg(const float* __restrict__ h, const float* __restrict__ asn,
           const float* __restrict__ adn, const int* __restrict__ rowptr,
           const int* __restrict__ csr, const float* __restrict__ bias,
           float* __restrict__ out, int n) {
    int node = blockIdx.x * 8 + (threadIdx.x >> 5);
    int lane = threadIdx.x & 31;
    if (node >= n) return;
    int beg = rowptr[node], end = rowptr[node + 1];
    float adv = adn[node];
    float4 acc = {0.f, 0.f, 0.f, 0.f};
    float m = -3.0e38f, s = 0.f;
    for (int j = beg; j < end; ++j) {
        int src = csr[j];
        float l = asn[src] + adv;
        l = l > 0.f ? l : 0.2f * l;
        float nm    = fmaxf(m, l);
        float scale = __expf(m - nm);
        float p     = __expf(l - nm);
        const float4 hv = *(const float4*)(h + (size_t)src * HID + lane * 4);
        acc.x = acc.x * scale + p * hv.x;
        acc.y = acc.y * scale + p * hv.y;
        acc.z = acc.z * scale + p * hv.z;
        acc.w = acc.w * scale + p * hv.w;
        s = s * scale + p;
        m = nm;
    }
    float inv = 1.0f / s;  // deg >= 1 (self loop)
    const float4 b4 = *(const float4*)(bias + lane * 4);
    float4 r;
    r.x = gelu_tanh(acc.x * inv + b4.x);
    r.y = gelu_tanh(acc.y * inv + b4.y);
    r.z = gelu_tanh(acc.z * inv + b4.z);
    r.w = gelu_tanh(acc.w * inv + b4.w);
    *(float4*)(out + (size_t)node * HID + lane * 4) = r;
}

// ---------------------------------------------------------------------------
// Pooling: g[batch[n]] += h[n]  (graph count 128, buffer is L2-resident)
// ---------------------------------------------------------------------------
__global__ void k_zero_f32(float* __restrict__ p, int n) {
    int i = blockIdx.x * blockDim.x + threadIdx.x;
    if (i < n) p[i] = 0.f;
}
__global__ void k_pool(const float* __restrict__ h, const int* __restrict__ batch,
                       float* __restrict__ g, int n) {
    int i = blockIdx.x * blockDim.x + threadIdx.x;
    if (i >= n * HID) return;
    int node = i >> 7, col = i & 127;
    atomicAdd(&g[(size_t)batch[node] * HID + col], h[i]);
}

// ---------------------------------------------------------------------------
// Host-side orchestration
// ---------------------------------------------------------------------------
extern "C" void kernel_launch(void* const* d_in, const int* in_sizes, int n_in,
                              void* d_out, int out_size, void* d_ws, size_t ws_size,
                              hipStream_t stream) {
    (void)in_sizes; (void)n_in; (void)out_size; (void)ws_size;
    const int N = GN, E = GE;
    const int K1 = INDIM + EMBDIM;  // 192

    const float* x     = (const float*)d_in[0];
    const int*   nidx  = (const int*)d_in[1];
    const int*   esrc  = (const int*)d_in[2];        // edge_index[0]
    const int*   edst  = (const int*)d_in[2] + E;    // edge_index[1]
    const int*   batch = (const int*)d_in[3];
    const float* emb   = (const float*)d_in[4];
    const float* W1  = (const float*)d_in[5];
    const float* a1s = (const float*)d_in[6];
    const float* a1d = (const float*)d_in[7];
    const float* b1  = (const float*)d_in[8];
    const float* W2  = (const float*)d_in[9];
    const float* a2s = (const float*)d_in[10];
    const float* a2d = (const float*)d_in[11];
    const float* b2  = (const float*)d_in[12];
    const float* W3  = (const float*)d_in[13];
    const float* a3s = (const float*)d_in[14];
    const float* a3d = (const float*)d_in[15];
    const float* b3  = (const float*)d_in[16];
    const float* fcW = (const float*)d_in[17];
    const float* fcb = (const float*)d_in[18];
    float* out = (float*)d_out;

    char* ws = (char*)d_ws;
    size_t off = 0;
    auto alloc = [&](size_t bytes) -> void* {
        void* p = ws + off;
        off = (off + bytes + 255) & ~(size_t)255;
        return p;
    };
    float* h0   = (float*)alloc((size_t)N * K1  * 4);
    float* P    = (float*)alloc((size_t)N * HID * 4);   // linear output
    float* Q    = (float*)alloc((size_t)N * HID * 4);   // layer output
    float* asn  = (float*)alloc((size_t)N * 4);
    float* adn  = (float*)alloc((size_t)N * 4);
    int*   cnt  = (int*)  alloc((size_t)N * 4);         // deg, then next[]
    int*   rptr = (int*)  alloc((size_t)(N + 1) * 4);
    int*   csr  = (int*)  alloc((size_t)(E + N) * 4);
    unsigned short* Wt1 = (unsigned short*)alloc((size_t)K1  * HID   * 2);
    unsigned short* Wt2 = (unsigned short*)alloc((size_t)HID * HID   * 2);
    unsigned short* Wt3 = (unsigned short*)alloc((size_t)HID * HID   * 2);
    unsigned short* WtF = (unsigned short*)alloc((size_t)HID * OUTDIM * 2);
    float* g = (float*)alloc((size_t)NGRAPH * HID * 4);

    // weight conversion (bf16, transposed)
    k_w_to_bf16t<<<(K1 * HID + 255) / 256, 256, 0, stream>>>(W1, Wt1, K1, HID);
    k_w_to_bf16t<<<(HID * HID + 255) / 256, 256, 0, stream>>>(W2, Wt2, HID, HID);
    k_w_to_bf16t<<<(HID * HID + 255) / 256, 256, 0, stream>>>(W3, Wt3, HID, HID);
    k_w_to_bf16t<<<(HID * OUTDIM + 255) / 256, 256, 0, stream>>>(fcW, WtF, HID, OUTDIM);

    // node features
    k_feat<<<N, 128, 0, stream>>>(x, nidx, emb, h0);

    // CSR by destination (self-loops included)
    k_cnt_init <<<(N + 255) / 256, 256, 0, stream>>>(cnt, N);
    k_cnt_edges<<<(E + 255) / 256, 256, 0, stream>>>(edst, cnt, E);
    k_scan<<<1, 1024, 0, stream>>>(cnt, rptr, cnt, N);
    k_fill_edges<<<(E + 255) / 256, 256, 0, stream>>>(esrc, edst, cnt, csr, E);
    k_fill_self <<<(N + 255) / 256, 256, 0, stream>>>(cnt, csr, N);

    const int gemmBlocks = (N + 127) / 128;
    const int nodeBlocks = (N + 7) / 8;

    // ---- layer 1 ----
    k_gemm_wmma<192, 8, 0><<<gemmBlocks, 256, 0, stream>>>(
        h0, (const unsigned int*)Wt1, nullptr, P, N);
    k_attvec<<<nodeBlocks, 256, 0, stream>>>(P, a1s, a1d, asn, adn, N);
    k_agg<<<nodeBlocks, 256, 0, stream>>>(P, asn, adn, rptr, csr, b1, Q, N);
    // ---- layer 2 ----
    k_gemm_wmma<128, 8, 0><<<gemmBlocks, 256, 0, stream>>>(
        Q, (const unsigned int*)Wt2, nullptr, P, N);
    k_attvec<<<nodeBlocks, 256, 0, stream>>>(P, a2s, a2d, asn, adn, N);
    k_agg<<<nodeBlocks, 256, 0, stream>>>(P, asn, adn, rptr, csr, b2, Q, N);
    // ---- layer 3 ----
    k_gemm_wmma<128, 8, 0><<<gemmBlocks, 256, 0, stream>>>(
        Q, (const unsigned int*)Wt3, nullptr, P, N);
    k_attvec<<<nodeBlocks, 256, 0, stream>>>(P, a3s, a3d, asn, adn, N);
    k_agg<<<nodeBlocks, 256, 0, stream>>>(P, asn, adn, rptr, csr, b3, Q, N);

    // ---- pool + FC (WMMA, fused bias + leaky_relu 0.01) ----
    k_zero_f32<<<(NGRAPH * HID + 255) / 256, 256, 0, stream>>>(g, NGRAPH * HID);
    k_pool<<<((size_t)N * HID + 255) / 256, 256, 0, stream>>>(Q, batch, g, N);
    k_gemm_wmma<128, 4, 2><<<1, 256, 0, stream>>>(
        g, (const unsigned int*)WtF, fcb, out, NGRAPH);
}
